// MambaBlock_38242388803943
// MI455X (gfx1250) — matrix operand order repaired and checked by reading, not compiled, not run on hardware
//
#include <hip/hip_runtime.h>
#include <cstdint>
#include <cstddef>

typedef unsigned short u16;
typedef __attribute__((ext_vector_type(16))) __bf16 v16bf;
typedef __attribute__((ext_vector_type(8)))  float  v8f;
typedef __attribute__((ext_vector_type(4)))  unsigned uint4x;

#define DEV __device__ __forceinline__

// ---------- scalar helpers ----------
DEV u16 f2bf(float f) {                       // RNE f32 -> bf16
  unsigned u = __float_as_uint(f);
  u += 0x7FFFu + ((u >> 16) & 1u);
  return (u16)(u >> 16);
}
DEV float bf2f(u16 h) { return __uint_as_float(((unsigned)h) << 16); }
DEV float sigmoidf_(float x) { return 1.f / (1.f + __expf(-x)); }
DEV float siluf_(float x) { return x * sigmoidf_(x); }
DEV float logsigf_(float x) { return fminf(x, 0.f) - log1pf(__expf(-fabsf(x))); }

union Frag { uint4x u[2]; v16bf v; };         // 32 bytes = 16 bf16

DEV v8f wmma_bf16(const Frag& a, const Frag& b, v8f c) {
  return __builtin_amdgcn_wmma_f32_16x16x32_bf16(
      false, a.v, false, b.v, (short)0, c, false, false);
}

// ---------- problem constants ----------
constexpr int kB  = 2;
constexpr int kL  = 4096;
constexpr int kD  = 2048;
constexpr int kH  = 32;
constexpr int kN  = 128;
constexpr int kP  = 64;      // D/H
constexpr int kT  = 64;      // block_len
constexpr int kC  = 64;      // L/block_len
constexpr int kTok  = kB * kL;                  // 8192
constexpr int kProj = kD + kH + 2 * kH * kN;    // 10272
constexpr int kProjPad = ((kProj + 63) / 64) * 64;  // 10304

// =====================================================================
// 1) depthwise causal conv(K=4) + silu gate -> bf16 x_mixed
// =====================================================================
__global__ __launch_bounds__(256) void conv_mix_k(
    const float* __restrict__ x, const float* __restrict__ cw,
    u16* __restrict__ xmix)
{
  long i = (long)blockIdx.x * 256 + threadIdx.x;        // over Tok*D
  if (i >= (long)kTok * kD) return;
  int d  = (int)(i % kD);
  long bl = i / kD;                                     // b*L + l
  int l  = (int)(bl % kL);
  float acc = 0.f;
#pragma unroll
  for (int k = 0; k < 4; ++k) {
    int ls = l + k - 3;
    float xv = (ls >= 0) ? x[(bl + k - 3) * kD + d] : 0.f;
    acc = fmaf(cw[d * 4 + k], xv, acc);
  }
  xmix[i] = f2bf(siluf_(acc) * x[bl * kD + d]);
}

// =====================================================================
// 2) generic f32 -> bf16 conversion
// =====================================================================
__global__ __launch_bounds__(256) void cvt_bf16_k(
    const float* __restrict__ src, u16* __restrict__ dst, long n)
{
  long i = (long)blockIdx.x * 256 + threadIdx.x;
  if (i < n) dst[i] = f2bf(src[i]);
}

// =====================================================================
// 3) WMMA bf16 GEMM: out[m,n] = sum_k A[m,k]*W[n,k]
//    BM=128, BN=64, BK=64; 256 threads = 8 waves (4M x 2N), wave tile 32x32
//    -> 8 v_wmma per barrier pair, A/B fragment reuse x2.
//    EPI 0: proj split+activation | 1: gate (Y *= sigmoid) | 2: out (+resid)
// =====================================================================
template<int EPI>
__global__ __launch_bounds__(256) void gemm_bf16_k(
    const u16* __restrict__ A, const u16* __restrict__ W, int K, int Nact,
    u16* __restrict__ xb, float* __restrict__ ab,
    u16* __restrict__ bvb, u16* __restrict__ cvb,
    float* __restrict__ ybuf,
    const float* __restrict__ resid, float* __restrict__ outp)
{
  __shared__ __align__(16) u16 As[128 * 64];   // 16 KB
  __shared__ __align__(16) u16 Ws[64 * 64];    //  8 KB
  const int tid  = threadIdx.x;
  const int lane = tid & 31, wv = tid >> 5;
  const int wvM  = wv >> 1, wvN = wv & 1;
  const int half = lane >> 4, l16 = lane & 15;
  const long gm0 = (long)blockIdx.y * 128;
  const long gn0 = (long)blockIdx.x * 64;
  v8f acc[2][2] = {};

  const int arow = tid >> 1, acol = (tid & 1) * 32;       // 128 rows, 2 thr/row
  const u16* srcA = A + (gm0 + arow) * (long)K + acol;
  const int wrow = (tid & 127) >> 1, wcol = (tid & 1) * 32;
  const u16* srcW = W + (gn0 + wrow) * (long)K + wcol;

  for (int k0 = 0; k0 < K; k0 += 64) {
    {
      const uint4x* s = (const uint4x*)(srcA + k0);
      uint4x* d = (uint4x*)&As[arow * 64 + acol];
      d[0] = s[0]; d[1] = s[1]; d[2] = s[2]; d[3] = s[3];
    }
    if (tid < 128) {
      const uint4x* s = (const uint4x*)(srcW + k0);
      uint4x* d = (uint4x*)&Ws[wrow * 64 + wcol];
      d[0] = s[0]; d[1] = s[1]; d[2] = s[2]; d[3] = s[3];
    }
    if (k0 + 64 < K) {
      __builtin_prefetch(srcA + k0 + 64, 0, 3);   // near-scope prefetch
      __builtin_prefetch(srcW + k0 + 64, 0, 3);
    }
    __syncthreads();

#pragma unroll
    for (int ks = 0; ks < 2; ++ks) {
      const int kc = ks * 32;
      // A fragments: lane row m=lane%16; K = {half*8..+8} U {16+half*8..+8}
      Frag a0, a1, b0, b1;
      a0.u[0] = *(const uint4x*)&As[(wvM * 32 + l16) * 64 + kc + half * 8];
      a0.u[1] = *(const uint4x*)&As[(wvM * 32 + l16) * 64 + kc + 16 + half * 8];
      a1.u[0] = *(const uint4x*)&As[(wvM * 32 + 16 + l16) * 64 + kc + half * 8];
      a1.u[1] = *(const uint4x*)&As[(wvM * 32 + 16 + l16) * 64 + kc + 16 + half * 8];
      // B fragments: lane col n=lane%16; K = {half*8..+8} U {16+half*8..+8}
      b0.u[0] = *(const uint4x*)&Ws[(wvN * 32 + l16) * 64 + kc + half * 8];
      b0.u[1] = *(const uint4x*)&Ws[(wvN * 32 + l16) * 64 + kc + 16 + half * 8];
      b1.u[0] = *(const uint4x*)&Ws[(wvN * 32 + 16 + l16) * 64 + kc + half * 8];
      b1.u[1] = *(const uint4x*)&Ws[(wvN * 32 + 16 + l16) * 64 + kc + 16 + half * 8];

      acc[0][0] = wmma_bf16(a0, b0, acc[0][0]);
      acc[0][1] = wmma_bf16(a0, b1, acc[0][1]);
      acc[1][0] = wmma_bf16(a1, b0, acc[1][0]);
      acc[1][1] = wmma_bf16(a1, b1, acc[1][1]);
    }
    __syncthreads();
  }

  // D layout: lane col n=lane%16; VGPR r -> row r + 8*half
#pragma unroll
  for (int msub = 0; msub < 2; ++msub) {
#pragma unroll
    for (int nsub = 0; nsub < 2; ++nsub) {
      const long n = gn0 + wvN * 32 + nsub * 16 + l16;
      if (n >= Nact) continue;                  // pad columns (proj only)
#pragma unroll
      for (int vg = 0; vg < 8; ++vg) {
        const long m = gm0 + wvM * 32 + msub * 16 + half * 8 + vg;
        float v = acc[msub][nsub][vg];
        if (EPI == 0) {
          if (n < kD)               xb[m * kD + n] = f2bf(siluf_(v));
          else if (n < kD + kH)     ab[m * kH + (n - kD)] = logsigf_(v);
          else if (n < kD + kH + kH * kN)
                                    bvb[m * (kH * kN) + (n - kD - kH)] = f2bf(siluf_(v));
          else                      cvb[m * (kH * kN) + (n - kD - kH - kH * kN)] = f2bf(siluf_(v));
        } else if (EPI == 1) {
          ybuf[m * kD + n] *= sigmoidf_(v);
        } else {
          outp[m * kD + n] = v + resid[m * kD + n];
        }
      }
    }
  }
}

// =====================================================================
// 4) A cumsum: Abuf[token,h] -> Acum[b,h,c,t] (cumsum over t)
// =====================================================================
__global__ __launch_bounds__(256) void acum_k(
    const float* __restrict__ Ab, float* __restrict__ Acum)
{
  int i = blockIdx.x * 256 + threadIdx.x;       // over B*H*C = 4096
  if (i >= kB * kH * kC) return;
  int c = i % kC, bh = i / kC;
  int h = bh % kH, b = bh / kH;
  float s = 0.f;
  for (int t = 0; t < kT; ++t) {
    s += Ab[((long)(b * kL + c * kT + t)) * kH + h];
    Acum[(((long)bh) * kC + c) * kT + t] = s;
  }
}

// =====================================================================
// 5) per (b,c,h): Y_diag and chunk states (f32 VALU; <64KB LDS)
// =====================================================================
__global__ __launch_bounds__(256) void ydiag_states_k(
    const u16* __restrict__ Bv, const u16* __restrict__ Cv,
    const u16* __restrict__ Xb, const float* __restrict__ Acum,
    float* __restrict__ Ybuf, float* __restrict__ states)
{
  __shared__ u16  Csh[64][128];
  __shared__ u16  Bsh[64][128];
  __shared__ u16  Xsh[64][64];
  __shared__ float Gs[64][64];
  __shared__ float acs[64];
  __shared__ float dec[64];
  const int blk = blockIdx.x;                      // ((b*C)+c)*H + h
  const int h = blk % kH, c = (blk / kH) % kC, b = blk / (kH * kC);
  const long tok0 = (long)b * kL + c * kT;
  const int tid = threadIdx.x;

  for (int o = tid; o < 64 * 128; o += 256) {
    int l = o >> 7, n = o & 127;
    Csh[l][n] = Cv[(tok0 + l) * (kH * kN) + h * kN + n];
    Bsh[l][n] = Bv[(tok0 + l) * (kH * kN) + h * kN + n];
  }
  for (int o = tid; o < 64 * 64; o += 256) {
    int l = o >> 6, p = o & 63;
    Xsh[l][p] = Xb[(tok0 + l) * kD + h * kP + p];
  }
  if (tid < 64) acs[tid] = Acum[(((long)(b * kH + h)) * kC + c) * kT + tid];
  __syncthreads();
  if (tid < 64) dec[tid] = __expf(acs[63] - acs[tid]);

  // G[l,s] = exp(acs[l]-acs[s]) * <C_l, B_s>   (lower triangular)
  for (int o = tid; o < 4096; o += 256) {
    int l = o >> 6, s = o & 63;
    float g = 0.f;
    if (s <= l) {
      float dot = 0.f;
      for (int n = 0; n < kN; ++n) dot = fmaf(bf2f(Csh[l][n]), bf2f(Bsh[s][n]), dot);
      g = dot * __expf(acs[l] - acs[s]);
    }
    Gs[l][s] = g;
  }
  __syncthreads();

  // Y_diag[l,p] = sum_{s<=l} G[l,s] * X[s,p]
  for (int o = tid; o < 4096; o += 256) {
    int l = o >> 6, p = o & 63;
    float y = 0.f;
    for (int s = 0; s <= l; ++s) y = fmaf(Gs[l][s], bf2f(Xsh[s][p]), y);
    Ybuf[(tok0 + l) * kD + h * kP + p] = y;
  }

  // states[p,n] = sum_l B[l,n]*exp(alast-acs[l])*X[l,p]
  for (int o = tid; o < kP * kN; o += 256) {
    int p = o >> 7, n = o & 127;
    float s = 0.f;
    for (int l = 0; l < kT; ++l)
      s = fmaf(bf2f(Bsh[l][n]) * dec[l], bf2f(Xsh[l][p]), s);
    states[(long)blk * (kP * kN) + o] = s;
  }
}

// =====================================================================
// 6) inter-chunk scan (in place): states[c] <- new_states[c]; final -> out
// =====================================================================
__global__ __launch_bounds__(256) void scan_k(
    const float* __restrict__ Acum, const float* __restrict__ init,
    float* __restrict__ states, float* __restrict__ finalst)
{
  __shared__ float ach[64];
  const int bh = blockIdx.x;                       // b*H + h
  const int b = bh >> 5, h = bh & 31;
  if (threadIdx.x < kC)
    ach[threadIdx.x] = __expf(Acum[(((long)bh) * kC + threadIdx.x) * kT + (kT - 1)]);
  __syncthreads();
  for (int e = threadIdx.x; e < kP * kN; e += 256) {
    float R = init[(long)bh * (kP * kN) + e];
    for (int c = 0; c < kC; ++c) {
      long idx = ((((long)b * kC + c) * kH + h) * (kP * kN)) + e;
      float S = states[idx];
      states[idx] = R;                              // inter_states[c]
      R = fmaf(ach[c], R, S);
    }
    finalst[(long)bh * (kP * kN) + e] = R;          // final_state
  }
}

// =====================================================================
// 7) per (b,c,h): Y_off[l,p] += exp(acs[l]) * <C_l, S[p,:]>
// =====================================================================
__global__ __launch_bounds__(256) void yoff_k(
    const u16* __restrict__ Cv, const float* __restrict__ states,
    const float* __restrict__ Acum, float* __restrict__ Ybuf)
{
  __shared__ u16   Csh[64][128];
  __shared__ float Ss[64][128];                    // [p][n]
  __shared__ float eac[64];
  const int blk = blockIdx.x;
  const int h = blk % kH, c = (blk / kH) % kC, b = blk / (kH * kC);
  const long tok0 = (long)b * kL + c * kT;
  const int tid = threadIdx.x;
  for (int o = tid; o < 64 * 128; o += 256) {
    int l = o >> 7, n = o & 127;
    Csh[l][n] = Cv[(tok0 + l) * (kH * kN) + h * kN + n];
    (&Ss[0][0])[o] = states[(long)blk * (kP * kN) + o];
  }
  if (tid < 64) eac[tid] = __expf(Acum[(((long)(b * kH + h)) * kC + c) * kT + tid]);
  __syncthreads();
  for (int o = tid; o < 4096; o += 256) {
    int l = o >> 6, p = o & 63;
    float dot = 0.f;
    for (int n = 0; n < kN; ++n) dot = fmaf(bf2f(Csh[l][n]), Ss[p][n], dot);
    Ybuf[(tok0 + l) * kD + h * kP + p] += dot * eac[l];
  }
}

// =====================================================================
// 9) RMSNorm over D, write bf16 (into reused x_mixed buffer)
// =====================================================================
__global__ __launch_bounds__(256) void rmsnorm_k(
    const float* __restrict__ Y, const float* __restrict__ nw,
    u16* __restrict__ yn)
{
  __shared__ float red[256];
  const long t = blockIdx.x;
  float s = 0.f;
  for (int d = threadIdx.x; d < kD; d += 256) {
    float v = Y[t * kD + d];
    s = fmaf(v, v, s);
  }
  red[threadIdx.x] = s;
  __syncthreads();
  for (int w = 128; w > 0; w >>= 1) {
    if (threadIdx.x < w) red[threadIdx.x] += red[threadIdx.x + w];
    __syncthreads();
  }
  const float rms = rsqrtf(red[0] / (float)kD + 1.1920929e-07f);
  for (int d = threadIdx.x; d < kD; d += 256)
    yn[t * kD + d] = f2bf(Y[t * kD + d] * rms * nw[d]);
}

// =====================================================================
// host launcher
// =====================================================================
extern "C" void kernel_launch(void* const* d_in, const int* in_sizes, int n_in,
                              void* d_out, int out_size, void* d_ws, size_t ws_size,
                              hipStream_t stream) {
  (void)in_sizes; (void)n_in; (void)out_size; (void)ws_size;
  const float* x   = (const float*)d_in[0];   // (B,L,D)
  const float* st0 = (const float*)d_in[1];   // (B,H,P,N)
  const float* Wp  = (const float*)d_in[2];   // (PROJ,D)
  const float* cw  = (const float*)d_in[3];   // (D,1,4)
  const float* gw  = (const float*)d_in[4];   // (D,D)
  const float* nw  = (const float*)d_in[5];   // (D,)
  const float* ow  = (const float*)d_in[6];   // (D,D)

  float* outy  = (float*)d_out;                       // (B,L,D)
  float* outst = outy + (size_t)kTok * kD;            // (B,H,P,N)

  char* w = (char*)d_ws;
  auto alloc = [&](size_t bytes) -> char* {
    char* p = w;
    w += (bytes + 255) & ~(size_t)255;
    return p;
  };
  u16*   xmixb  = (u16*)  alloc((size_t)kTok * kD * 2);
  u16*   Wpb    = (u16*)  alloc((size_t)kProjPad * kD * 2);   // padded rows
  u16*   gwb    = (u16*)  alloc((size_t)kD * kD * 2);
  u16*   owb    = (u16*)  alloc((size_t)kD * kD * 2);
  u16*   Xb     = (u16*)  alloc((size_t)kTok * kD * 2);
  float* Ab     = (float*)alloc((size_t)kTok * kH * 4);
  u16*   Bvb    = (u16*)  alloc((size_t)kTok * kH * kN * 2);
  u16*   Cvb    = (u16*)  alloc((size_t)kTok * kH * kN * 2);
  float* Acum   = (float*)alloc((size_t)kB * kH * kC * kT * 4);
  float* states = (float*)alloc((size_t)kB * kC * kH * kP * kN * 4);
  float* Ybuf   = (float*)alloc((size_t)kTok * kD * 4);

  // 1) conv + silu mix
  conv_mix_k<<<(int)(((long)kTok * kD + 255) / 256), 256, 0, stream>>>(x, cw, xmixb);

  // 2) weight conversions
  cvt_bf16_k<<<(int)(((long)kProj * kD + 255) / 256), 256, 0, stream>>>(Wp, Wpb, (long)kProj * kD);
  cvt_bf16_k<<<(int)(((long)kD * kD + 255) / 256), 256, 0, stream>>>(gw, gwb, (long)kD * kD);
  cvt_bf16_k<<<(int)(((long)kD * kD + 255) / 256), 256, 0, stream>>>(ow, owb, (long)kD * kD);

  // 3) projection GEMM with split/activation epilogue
  gemm_bf16_k<0><<<dim3(kProjPad / 64, kTok / 128), 256, 0, stream>>>(
      xmixb, Wpb, kD, kProj, Xb, Ab, Bvb, Cvb, nullptr, nullptr, nullptr);

  // 4) cumulative sums of log-decay
  acum_k<<<(kB * kH * kC + 255) / 256, 256, 0, stream>>>(Ab, Acum);

  // 5) intra-chunk Y and per-chunk states
  ydiag_states_k<<<kB * kC * kH, 256, 0, stream>>>(Bvb, Cvb, Xb, Acum, Ybuf, states);

  // 6) inter-chunk scan (writes final_state to output tail)
  scan_k<<<kB * kH, 256, 0, stream>>>(Acum, st0, states, outst);

  // 7) inter-chunk contribution to Y
  yoff_k<<<kB * kC * kH, 256, 0, stream>>>(Cvb, states, Acum, Ybuf);

  // 8) gate GEMM, Y *= sigmoid(gate)
  gemm_bf16_k<1><<<dim3(kD / 64, kTok / 128), 256, 0, stream>>>(
      xmixb, gwb, kD, kD, nullptr, nullptr, nullptr, nullptr, Ybuf, nullptr, nullptr);

  // 9) RMSNorm -> bf16 (reuse xmixb)
  rmsnorm_k<<<kTok, 256, 0, stream>>>(Ybuf, nw, xmixb);

  // 10) output GEMM + residual
  gemm_bf16_k<2><<<dim3(kD / 64, kTok / 128), 256, 0, stream>>>(
      xmixb, owb, kD, kD, nullptr, nullptr, nullptr, nullptr, nullptr, x, outy);
}
